// DeformableAttention_43997644981052
// MI455X (gfx1250) — matrix operand order repaired and checked by reading, not compiled, not run on hardware
//
#include <hip/hip_runtime.h>

// Problem constants (reference: D=256, NH=8, NL=3, NP=4, HD=32, B=4, Q=10000)
constexpr int kD  = 256;
constexpr int kNH = 8;
constexpr int kNL = 3;
constexpr int kNP = 4;
constexpr int kHD = 32;
constexpr int kB  = 4;
constexpr int kQ  = 10000;
constexpr int kHWsum = 13125;            // 10000 + 2500 + 625

typedef __attribute__((ext_vector_type(16))) __bf16 v16bf;
typedef __attribute__((ext_vector_type(8)))  __bf16 v8bf;
typedef __attribute__((ext_vector_type(4)))  __bf16 v4bf;
typedef __attribute__((ext_vector_type(8)))  float  v8f;

__device__ __forceinline__ __bf16 f2bf(float f) {
  unsigned u = __builtin_bit_cast(unsigned, f);
  unsigned r = u + 0x7FFFu + ((u >> 16) & 1u);   // round-to-nearest-even
  return __builtin_bit_cast(__bf16, (unsigned short)(r >> 16));
}
__device__ __forceinline__ float bf2f(__bf16 b) {
  unsigned u = (unsigned)__builtin_bit_cast(unsigned short, b) << 16;
  return __builtin_bit_cast(float, u);
}

// ---------------------------------------------------------------- converts --
__global__ __launch_bounds__(256) void cvt_f32_bf16(
    const float* __restrict__ in, __bf16* __restrict__ out, int n)
{
  const int i = (int)(blockIdx.x * blockDim.x + threadIdx.x) * 4;
  if (i + 3 < n) {
    const float4 v = *reinterpret_cast<const float4*>(in + i);
    v4bf o; o[0] = f2bf(v.x); o[1] = f2bf(v.y); o[2] = f2bf(v.z); o[3] = f2bf(v.w);
    *reinterpret_cast<v4bf*>(out + i) = o;
  } else {
    for (int k = 0; k < 4 && i + k < n; ++k) out[i + k] = f2bf(in[i + k]);
  }
}

// in [R,C] f32  ->  out [C,R] bf16   (LDS 32x32 tile transpose)
__global__ __launch_bounds__(256) void transpose_cvt_bf16(
    const float* __restrict__ in, __bf16* __restrict__ out, int R, int C)
{
  __shared__ float tile[32][33];
  const int tx = threadIdx.x, ty = threadIdx.y;          // blockDim (32,8)
  const int cx = blockIdx.x * 32 + tx;
  const int r0 = blockIdx.y * 32;
#pragma unroll
  for (int j = 0; j < 32; j += 8) {
    const int r = r0 + ty + j;
    if (cx < C && r < R) tile[ty + j][tx] = in[(size_t)r * C + cx];
  }
  __syncthreads();
  const int ox = r0 + tx;                                // out col = in row
  const int o0 = blockIdx.x * 32;                        // out row = in col
#pragma unroll
  for (int j = 0; j < 32; j += 8) {
    const int oy = o0 + ty + j;
    if (ox < R && oy < C) out[(size_t)oy * R + ox] = f2bf(tile[tx][ty + j]);
  }
}

// ------------------------------------------------------------------- GEMM --
// C[M,N] = A * W^T + bias with v_wmma_f32_16x16x32_bf16, K fixed = 256.
//   A  : [M,256] bf16 row-major
//   Wt : [N,256] bf16 row-major (pre-transposed weights)
// Every fragment operand is two contiguous 16B loads per lane (b128 path):
//   A 16x32 layout: lane half 0 -> K 0..7 / 16..23 ; half 1 -> K 8..15 / 24..31
//   B 32x16 layout: lane half 0 -> K 0..7 / 16..23 ; half 1 -> K 8..15 / 24..31
// Block = 256 threads = 8 waves arranged 4(M) x 2(N) -> 64x32 tile.
template <bool OUTBF>
__global__ __launch_bounds__(256) void wmma_gemm(
    const __bf16* __restrict__ A, const __bf16* __restrict__ Wt,
    const float* __restrict__ bias,
    float* __restrict__ Cf, __bf16* __restrict__ Cb, int M, int N)
{
  constexpr int K = kD;
  const int lane = threadIdx.x & 31;
  const int wave = threadIdx.x >> 5;
  const int m0 = blockIdx.y * 64 + (wave & 3) * 16;
  const int n0 = blockIdx.x * 32 + (wave >> 2) * 16;
  if (m0 >= M) return;                                   // wave-uniform exit

  const int half = lane >> 4;
  const int l15  = lane & 15;
  int mIdx = m0 + l15; if (mIdx >= M) mIdx = M - 1;      // clamp; stores guarded
  const int nIdx = n0 + l15;
  const __bf16* Arow = A  + (size_t)mIdx * K;
  const __bf16* Wrow = Wt + (size_t)nIdx * K;

  v8f acc = {};
#pragma unroll
  for (int k0 = 0; k0 < K; k0 += 32) {
    const v8bf alo = *reinterpret_cast<const v8bf*>(Arow + k0 + (half << 3));
    const v8bf ahi = *reinterpret_cast<const v8bf*>(Arow + k0 + 16 + (half << 3));
    const v8bf blo = *reinterpret_cast<const v8bf*>(Wrow + k0 + (half << 3));
    const v8bf bhi = *reinterpret_cast<const v8bf*>(Wrow + k0 + 16 + (half << 3));
    const v16bf a = __builtin_shufflevector(alo, ahi, 0,1,2,3,4,5,6,7,8,9,10,11,12,13,14,15);
    const v16bf b = __builtin_shufflevector(blo, bhi, 0,1,2,3,4,5,6,7,8,9,10,11,12,13,14,15);
    acc = __builtin_amdgcn_wmma_f32_16x16x32_bf16(false, a, false, b,
                                                  (short)0, acc, false, false);
  }

  const float bn = bias[nIdx];
#pragma unroll
  for (int r = 0; r < 8; ++r) {
    const int m = m0 + r + (half << 3);                  // C: VGPR r -> row r / r+8
    if (m < M) {
      const float v = acc[r] + bn;
      if constexpr (OUTBF) Cb[(size_t)m * N + nIdx] = f2bf(v);
      else                 Cf[(size_t)m * N + nIdx] = v;
    }
  }
}

// -------------------------------------------------------------- sampling --
// One wave32 per (b, q, head); lane = head channel (HD == 32).
// vProj bf16 [B][HWsum][D] channel-last: each corner gather is one coalesced
// 64B wave load; the whole 26.9MB tensor is L2-resident (192MB L2).
__global__ __launch_bounds__(256) void deform_sample(
    const __bf16* __restrict__ vProj,    // [B][13125][256] bf16
    const float*  __restrict__ refp,     // [B*Q][2]
    const float*  __restrict__ offsRaw,  // [B*Q][192]  (pre-tanh logits)
    const float*  __restrict__ attnRaw,  // [B*Q][96]   (pre-softmax logits)
    __bf16* __restrict__ sampOut)        // [B*Q][256] bf16
{
  const int gw   = (int)((blockIdx.x * blockDim.x + threadIdx.x) >> 5);
  const int lane = threadIdx.x & 31;
  if (gw >= kB * kQ * kNH) return;
  const int h  = gw & (kNH - 1);
  const int bq = gw >> 3;
  const int b  = bq / kQ;

  const float rx = refp[(size_t)bq * 2 + 0];
  const float ry = refp[(size_t)bq * 2 + 1];

  // softmax over the 12 (level,point) logits of this head (broadcast loads)
  const float* ar = attnRaw + (size_t)bq * (kNH * kNL * kNP) + h * (kNL * kNP);
  float aw[kNL * kNP];
  float mx = -3.4e38f;
#pragma unroll
  for (int i = 0; i < kNL * kNP; ++i) { aw[i] = ar[i]; mx = fmaxf(mx, aw[i]); }
  float s = 0.f;
#pragma unroll
  for (int i = 0; i < kNL * kNP; ++i) { aw[i] = __expf(aw[i] - mx); s += aw[i]; }
  const float inv = 1.f / s;

  const float* orow = offsRaw + (size_t)bq * (kNH * kNL * kNP * 2) + h * (kNL * kNP * 2);

  const int Hs[kNL]  = {100, 50, 25};
  const int Wd[kNL]  = {100, 50, 25};
  const int Ofs[kNL] = {0, 10000, 12500};

  float acc = 0.f;
#pragma unroll
  for (int lvl = 0; lvl < kNL; ++lvl) {
    const int Hl = Hs[lvl], Wl = Wd[lvl];
    const __bf16* vb = vProj + ((size_t)b * kHWsum + Ofs[lvl]) * kD + h * kHD + lane;
#pragma unroll
    for (int p = 0; p < kNP; ++p) {
      // tanh(raw)*0.5 offset; grid_sample align_corners=False: x = loc*W - 0.5
      const float ox = 0.5f * tanhf(orow[lvl * 8 + p * 2 + 0]);
      const float oy = 0.5f * tanhf(orow[lvl * 8 + p * 2 + 1]);
      const float x = (rx + ox) * (float)Wl - 0.5f;
      const float y = (ry + oy) * (float)Hl - 0.5f;
      const float x0f = floorf(x), y0f = floorf(y);
      const float wx = x - x0f, wy = y - y0f;
      const int x0 = (int)x0f, y0 = (int)y0f;
      const int x1 = x0 + 1,   y1 = y0 + 1;
      const float aWt = aw[lvl * kNP + p] * inv;
      const float w00 = aWt * (1.f - wx) * (1.f - wy);
      const float w10 = aWt * wx * (1.f - wy);
      const float w01 = aWt * (1.f - wx) * wy;
      const float w11 = aWt * wx * wy;
      const bool vx0 = (x0 >= 0) & (x0 < Wl), vx1 = (x1 >= 0) & (x1 < Wl);
      const bool vy0 = (y0 >= 0) & (y0 < Hl), vy1 = (y1 >= 0) & (y1 < Hl);
      if (vx0 & vy0) acc += w00 * bf2f(vb[(size_t)(y0 * Wl + x0) * kD]);
      if (vx1 & vy0) acc += w10 * bf2f(vb[(size_t)(y0 * Wl + x1) * kD]);
      if (vx0 & vy1) acc += w01 * bf2f(vb[(size_t)(y1 * Wl + x0) * kD]);
      if (vx1 & vy1) acc += w11 * bf2f(vb[(size_t)(y1 * Wl + x1) * kD]);
    }
  }
  sampOut[(size_t)bq * kD + h * kHD + lane] = f2bf(acc);
}

// ------------------------------------------------------------------ launch --
extern "C" void kernel_launch(void* const* d_in, const int* in_sizes, int n_in,
                              void* d_out, int out_size, void* d_ws, size_t ws_size,
                              hipStream_t stream)
{
  (void)in_sizes; (void)n_in; (void)out_size; (void)ws_size;
  const float* query    = (const float*)d_in[0];
  const float* value[3] = {(const float*)d_in[1], (const float*)d_in[2],
                           (const float*)d_in[3]};
  const float* refp   = (const float*)d_in[4];
  const float* W_off  = (const float*)d_in[5];
  const float* b_off  = (const float*)d_in[6];
  const float* W_attn = (const float*)d_in[7];
  const float* b_attn = (const float*)d_in[8];
  const float* W_val  = (const float*)d_in[9];
  const float* b_val  = (const float*)d_in[10];
  const float* W_out  = (const float*)d_in[11];
  const float* b_out  = (const float*)d_in[12];
  float* out = (float*)d_out;

  const int M = kB * kQ;                                 // 40000 rows
  // workspace layout (f32 region first, then bf16 region; all 16B aligned)
  float*  offsRaw = (float*)d_ws;                        //  7,680,000 f32
  float*  attnRaw = offsRaw + (size_t)M * (kNH * kNL * kNP * 2);  // 3,840,000 f32
  __bf16* qbf     = (__bf16*)(attnRaw + (size_t)M * (kNH * kNL * kNP));
  __bf16* WtOff   = qbf    + (size_t)M * kD;             // [192,256]
  __bf16* WtAttn  = WtOff  + 192 * kD;                   // [96,256]
  __bf16* WtVal   = WtAttn + 96 * kD;                    // [256,256]
  __bf16* WtOut   = WtVal  + kD * kD;                    // [256,256]
  __bf16* featT   = WtOut  + kD * kD;                    // [B*13125,256]
  __bf16* vProj   = featT  + (size_t)kB * kHWsum * kD;   // [B*13125,256]
  __bf16* sampOut = vProj  + (size_t)kB * kHWsum * kD;   // [40000,256]

  dim3 blk(256);
  dim3 tblk(32, 8);

  // 0) one-time bf16 packing: query + transposed weights + transposed features
  const int nq = M * kD;                                 // 10,240,000
  cvt_f32_bf16<<<dim3(nq / 4 / 256), blk, 0, stream>>>(query, qbf, nq);
  transpose_cvt_bf16<<<dim3((192 + 31) / 32, 8), tblk, 0, stream>>>(W_off,  WtOff,  kD, 192);
  transpose_cvt_bf16<<<dim3((96  + 31) / 32, 8), tblk, 0, stream>>>(W_attn, WtAttn, kD, 96);
  transpose_cvt_bf16<<<dim3(8, 8), tblk, 0, stream>>>(W_val, WtVal, kD, kD);
  transpose_cvt_bf16<<<dim3(8, 8), tblk, 0, stream>>>(W_out, WtOut, kD, kD);

  const int HWs[3] = {10000, 2500, 625};
  const int Ofs[3] = {0, 10000, 12500};
  for (int bb = 0; bb < kB; ++bb)
    for (int l = 0; l < 3; ++l) {
      const int HW = HWs[l];
      transpose_cvt_bf16<<<dim3((HW + 31) / 32, 8), tblk, 0, stream>>>(
          value[l] + (size_t)bb * kD * HW,
          featT + ((size_t)bb * kHWsum + Ofs[l]) * kD, kD, HW);
    }

  // 1) sampling-offset + attention-logit GEMMs (f32 outputs)
  wmma_gemm<false><<<dim3(192 / 32, M / 64), blk, 0, stream>>>(
      qbf, WtOff, b_off, offsRaw, nullptr, M, 192);
  wmma_gemm<false><<<dim3(96 / 32, M / 64), blk, 0, stream>>>(
      qbf, WtAttn, b_attn, attnRaw, nullptr, M, 96);

  // 2) value projection per (batch, level) -> bf16 channel-last vProj
  for (int bb = 0; bb < kB; ++bb)
    for (int l = 0; l < 3; ++l) {
      const int HW = HWs[l];
      wmma_gemm<true><<<dim3(kD / 32, (HW + 63) / 64), blk, 0, stream>>>(
          featT + ((size_t)bb * kHWsum + Ofs[l]) * kD, WtVal, b_val,
          nullptr, vProj + ((size_t)bb * kHWsum + Ofs[l]) * kD, HW, kD);
    }

  // 3) deformable bilinear sampling + attention-weighted sum (bf16 in/out)
  deform_sample<<<dim3((kB * kQ * kNH * 32) / 256), blk, 0, stream>>>(
      vProj, refp, offsRaw, attnRaw, sampOut);

  // 4) output projection -> final fp32 output
  wmma_gemm<false><<<dim3(kD / 32, M / 64), blk, 0, stream>>>(
      sampOut, WtOut, b_out, out, nullptr, M, kD);
}
